// SelfAttnPropagation_36498632081957
// MI455X (gfx1250) — matrix operand order repaired and checked by reading, not compiled, not run on hardware
//
#include <hip/hip_runtime.h>
#include <hip/hip_bf16.h>

// ---------------------------------------------------------------------------
// SelfAttnPropagation for MI455X (gfx1250): flash-attention on v_wmma f16,
// with async global->LDS double-buffered K/V staging (ASYNCcnt).
//   b=2, c=128, h=80, w=128, n=10240
//   q = x@Wq^T+bq ; k = q@Wk^T+bk ; v = q@Wv^T+bv   (k,v from q: source quirk)
//   P = softmax(q k^T / sqrt(c)) ; out = P @ [v | f0 | f1]
// ---------------------------------------------------------------------------

typedef _Float16 half_t;
typedef __attribute__((ext_vector_type(16))) _Float16 v16h;
typedef __attribute__((ext_vector_type(8)))  float    v8f;

#define B_      2
#define C_      128
#define H_      80
#define W_      128
#define N_      (H_ * W_)     // 10240
#define VW_     144           // v(128) + f0(2) + f1(2) + pad(12): 9 N-tiles
#define KT_     64            // key-chunk length
#define QR_     64            // rows per block (4 waves x 16 rows)
#define NCHUNK_ (N_ / KT_)    // 160

// --- gfx1250 async global->LDS path (guarded; falls back to sync copies) ---
#if defined(__gfx1250__) && __has_builtin(__builtin_amdgcn_global_load_async_to_lds_b128)
#define SAP_ASYNC 1
#else
#define SAP_ASYNC 0
#endif

#if SAP_ASYNC
// Builtin signature (from hipcc diagnostic): param0 is a pointer to
// int __attribute__((vector_size(16))) in the __device__ (AS1) language AS;
// param1 is the __shared__ (AS3) analogue. Build those exact pointer types.
typedef int sap_v4i __attribute__((vector_size(16)));
typedef __attribute__((address_space(1))) sap_v4i sap_gv4i;
typedef __attribute__((address_space(3))) sap_v4i sap_lv4i;
// Flat LDS addresses carry the LDS offset in addr[31:0] (aperture spec), and
// flat global addresses equal the global address, so integer round-trips give
// well-defined AS1/AS3 pointers without generic->ASn casts.
__device__ __forceinline__ void sap_async_b128(const void* g, void* l) {
  __builtin_amdgcn_global_load_async_to_lds_b128(
      (sap_gv4i*)(unsigned long long)(size_t)g,
      (sap_lv4i*)(unsigned)(size_t)l, 0, 0);
}
#if __has_builtin(__builtin_amdgcn_s_wait_asynccnt)
#define SAP_WAIT_ASYNC(n) __builtin_amdgcn_s_wait_asynccnt(n)
#else
#define SAP_WAIT_ASYNC(n) asm volatile("s_wait_asynccnt %0" ::"i"(n) : "memory")
#endif
#endif

__device__ __forceinline__ v8f wmma16(v16h a, v16h b, v8f c) {
  return __builtin_amdgcn_wmma_f32_16x16x32_f16(
      /*neg_a=*/false, a, /*neg_b=*/false, b,
      /*c_mod=*/(short)0, c, /*reuse_a=*/false, /*reuse_b=*/false);
}

union AF { v16h v; unsigned u[8]; };

// A fragment (16x32 f16, M x K) from row-major [M][K] source (ld halfs).
__device__ __forceinline__ v16h load_frag_a(const half_t* base, int ld) {
  const int lane = threadIdx.x & 31;
  const int m    = lane & 15;
  const int koff = (lane >> 4) * 16;
  const half_t* r = base + m * ld;
  AF f;
#pragma unroll
  for (int j = 0; j < 8; ++j) {
    f.u[j]     = *(const unsigned*)(r + koff + 2 * j);
  }
  return f.v;
}

// B fragment (32x16 f16, K x N) where element (k,n) lives at base[n*ld + k].
__device__ __forceinline__ v16h load_frag_b(const half_t* base, int ld) {
  const int lane = threadIdx.x & 31;
  const int n  = lane & 15;
  const int kb = (lane >> 4) * 16;
  const half_t* r = base + n * ld + kb;
  AF f;
#pragma unroll
  for (int j = 0; j < 8; ++j)
    f.u[j] = *(const unsigned*)(r + 2 * j);
  return f.v;
}

// ---------------------------------------------------------------------------
// Kernel 0: f32 -> f16 weight conversion (Wq, Wk, Wv), row-major [out][in]
// ---------------------------------------------------------------------------
__global__ __launch_bounds__(256) void sap_wconv(const float* __restrict__ Wq,
                                                 const float* __restrict__ Wk,
                                                 const float* __restrict__ Wv,
                                                 half_t* __restrict__ wh) {
  int idx = blockIdx.x * 256 + threadIdx.x;
  if (idx >= 3 * C_ * C_) return;
  int wsel = idx >> 14;
  int off  = idx & (C_ * C_ - 1);
  const float* src = (wsel == 0) ? Wq : (wsel == 1) ? Wk : Wv;
  wh[wsel * C_ * C_ + off] = (half_t)src[off];
}

// ---------------------------------------------------------------------------
// Kernel 1: projections. Emits f16 qh[b][n][128], kh[b][n][128], and V
// pre-transposed per key-chunk: vxT[b][chunk][144][64] = [v|f0|f1|0]^T so the
// flash kernel's V stage is a contiguous 18KB block copy.
// ---------------------------------------------------------------------------
__global__ __launch_bounds__(128) void sap_proj(
    const float* __restrict__ feat, const float* __restrict__ flow0,
    const float* __restrict__ flow1, const half_t* __restrict__ wh,
    const float* __restrict__ bq, const float* __restrict__ bk,
    const float* __restrict__ bv, half_t* __restrict__ qh,
    half_t* __restrict__ kh, half_t* __restrict__ vx) {
  __shared__ __align__(16) half_t xl[QR_ * C_];   // x tile [64][128]
  __shared__ __align__(16) half_t ql[QR_ * C_];   // q tile (A-operand staging)

  const int tid  = threadIdx.x;
  const int wv   = tid >> 5;
  const int lane = tid & 31;
  const int nlo  = lane & 15;
  const int hi   = lane >> 4;
  const int bidx = blockIdx.y;
  const int nb   = blockIdx.x * QR_;   // == chunk * KT_
  half_t* vb = vx + (size_t)(bidx * NCHUNK_ + blockIdx.x) * VW_ * KT_;

  // x[r][c] = feat[b][c][nb+r]
  for (int idx = tid; idx < QR_ * C_; idx += 128) {
    int c = idx >> 6;
    int r = idx & 63;
    xl[r * C_ + c] = (half_t)feat[((size_t)bidx * C_ + c) * N_ + nb + r];
  }
  __syncthreads();

  const half_t* whq = wh;
  const half_t* whk = wh + C_ * C_;
  const half_t* whv = wh + 2 * C_ * C_;

  v16h xa[4];
#pragma unroll
  for (int ks = 0; ks < 4; ++ks)
    xa[ks] = load_frag_a(xl + wv * 16 * C_ + ks * 32, C_);

  half_t* myql = ql + wv * 16 * C_;
#pragma unroll
  for (int nt = 0; nt < 8; ++nt) {
    v8f d = {};
#pragma unroll
    for (int ks = 0; ks < 4; ++ks)
      d = wmma16(xa[ks], load_frag_b(whq + nt * 16 * C_ + ks * 32, C_), d);
    float bias = bq[nt * 16 + nlo];
#pragma unroll
    for (int r = 0; r < 8; ++r) {
      float qv  = d[r] + bias;
      int mrow  = r + 8 * hi;
      int n     = nb + wv * 16 + mrow;
      qh[((size_t)bidx * N_ + n) * C_ + nt * 16 + nlo] = (half_t)qv;
      myql[mrow * C_ + nt * 16 + nlo] = (half_t)qv;
    }
  }
  __builtin_amdgcn_wave_barrier();   // per-wave LDS region; DS is in-order

  v16h qa[4];
#pragma unroll
  for (int ks = 0; ks < 4; ++ks)
    qa[ks] = load_frag_a(myql + ks * 32, C_);

#pragma unroll
  for (int nt = 0; nt < 8; ++nt) {
    v8f dk = {}, dv = {};
#pragma unroll
    for (int ks = 0; ks < 4; ++ks) {
      dk = wmma16(qa[ks], load_frag_b(whk + nt * 16 * C_ + ks * 32, C_), dk);
      dv = wmma16(qa[ks], load_frag_b(whv + nt * 16 * C_ + ks * 32, C_), dv);
    }
    float biask = bk[nt * 16 + nlo];
    float biasv = bv[nt * 16 + nlo];
#pragma unroll
    for (int r = 0; r < 8; ++r) {
      int mrow = r + 8 * hi;
      int kloc = wv * 16 + mrow;                 // n & 63
      int n    = nb + kloc;
      kh[((size_t)bidx * N_ + n) * C_ + nt * 16 + nlo] = (half_t)(dk[r] + biask);
      vb[(nt * 16 + nlo) * KT_ + kloc]                 = (half_t)(dv[r] + biasv);
    }
  }

  // flow channels + zero pad into the transposed V chunk
  if (lane < 16) {
    int kloc = wv * 16 + lane;
    int n    = nb + kloc;
    vb[128 * KT_ + kloc] = (half_t)flow0[((size_t)bidx * 2 + 0) * N_ + n];
    vb[129 * KT_ + kloc] = (half_t)flow0[((size_t)bidx * 2 + 1) * N_ + n];
    vb[130 * KT_ + kloc] = (half_t)flow1[((size_t)bidx * 2 + 0) * N_ + n];
    vb[131 * KT_ + kloc] = (half_t)flow1[((size_t)bidx * 2 + 1) * N_ + n];
#pragma unroll
    for (int j = 132; j < VW_; ++j) vb[j * KT_ + kloc] = (half_t)0.f;
  }
}

// ---------------------------------------------------------------------------
// Kernel 2: flash attention. 4 waves/block, 16 query rows/wave, 64-key
// chunks, double-buffered K/V in LDS via async global->LDS when available.
// ---------------------------------------------------------------------------
__global__ __launch_bounds__(128) void sap_flash(
    const half_t* __restrict__ qh, const half_t* __restrict__ kh,
    const half_t* __restrict__ vx, float* __restrict__ out) {
  __shared__ __align__(16) half_t kbuf[2][KT_ * C_];   // [key][c]   2x16 KB
  __shared__ __align__(16) half_t vbuf[2][VW_ * KT_];  // [c][key]   2x18 KB
  __shared__ __align__(16) half_t pbuf[4 * 16 * KT_];  // P staging     8 KB

  const int tid   = threadIdx.x;
  const int wv    = tid >> 5;
  const int lane  = tid & 31;
  const int nlo   = lane & 15;
  const int hi    = lane >> 4;
  const int bidx  = blockIdx.y;
  const int qbase = blockIdx.x * QR_ + wv * 16;
  const float sc  = 0.08838834764831845f;   // 1/sqrt(128)

  // chunk loader: K is contiguous [64][128] halfs; V is contiguous [144][64]
  auto issue_chunk = [&](int chunk, half_t* kdst, half_t* vdst) {
    const char* ksrc = (const char*)(kh + ((size_t)bidx * N_ + chunk * KT_) * C_);
    const char* vsrc = (const char*)(vx + (size_t)(bidx * NCHUNK_ + chunk) * VW_ * KT_);
#if SAP_ASYNC
    char* kd = (char*)kdst;
    char* vd = (char*)vdst;
    for (int i = tid; i < (KT_ * C_ * 2) / 16; i += 128)      // 8 per thread
      sap_async_b128(ksrc + i * 16, kd + i * 16);
    for (int i = tid; i < (VW_ * KT_ * 2) / 16; i += 128)     // 9 per thread
      sap_async_b128(vsrc + i * 16, vd + i * 16);
#else
    const unsigned* ks = (const unsigned*)ksrc;
    const unsigned* vs = (const unsigned*)vsrc;
    unsigned* kd = (unsigned*)kdst;
    unsigned* vd = (unsigned*)vdst;
    for (int i = tid; i < KT_ * C_ / 2; i += 128) kd[i] = ks[i];
    for (int i = tid; i < VW_ * KT_ / 2; i += 128) vd[i] = vs[i];
#endif
  };

  // Q fragments live in VGPRs for the whole kernel
  v16h qa[4];
#pragma unroll
  for (int ks = 0; ks < 4; ++ks)
    qa[ks] = load_frag_a(qh + ((size_t)bidx * N_ + qbase) * C_ + ks * 32, C_);

  v8f acc[9];
#pragma unroll
  for (int j = 0; j < 9; ++j) acc[j] = (v8f){};
  float mI[8], lI[8];
#pragma unroll
  for (int r = 0; r < 8; ++r) { mI[r] = -__builtin_inff(); lI[r] = 0.f; }

  half_t* myp = pbuf + wv * 16 * KT_;

  issue_chunk(0, kbuf[0], vbuf[0]);   // preload chunk 0
  int stage = 0;

  for (int ci = 0; ci < NCHUNK_; ++ci) {
    // prefetch next chunk into the other buffer (its last reader finished
    // before the trailing barrier of the previous iteration)
    if (ci + 1 < NCHUNK_) issue_chunk(ci + 1, kbuf[stage ^ 1], vbuf[stage ^ 1]);
#if SAP_ASYNC
    // async loads complete in order: <=17 outstanding => current chunk landed
    if (ci + 1 < NCHUNK_) SAP_WAIT_ASYNC(17);
    else                  SAP_WAIT_ASYNC(0);
#endif
    __syncthreads();   // every wave passed its own wait -> chunk ci visible

    const half_t* kb = kbuf[stage];
    const half_t* vbT = vbuf[stage];

    // S = Q K^T : 4 N-tiles x 4 K-steps = 16 WMMA
    v8f s[4];
#pragma unroll
    for (int t = 0; t < 4; ++t) {
      v8f d = {};
#pragma unroll
      for (int ks = 0; ks < 4; ++ks)
        d = wmma16(qa[ks], load_frag_b(kb + t * 16 * C_ + ks * 32, C_), d);
      s[t] = d;
    }

    // online softmax (a row spans 16 lanes in C/D layout)
    float mx[8], alpha[8], rs[8];
#pragma unroll
    for (int r = 0; r < 8; ++r) {
      float v0 = fmaxf(fmaxf(s[0][r], s[1][r]), fmaxf(s[2][r], s[3][r]));
      v0 = fmaxf(v0, __shfl_xor(v0, 1, 32));
      v0 = fmaxf(v0, __shfl_xor(v0, 2, 32));
      v0 = fmaxf(v0, __shfl_xor(v0, 4, 32));
      v0 = fmaxf(v0, __shfl_xor(v0, 8, 32));
      mx[r] = v0 * sc;
      float mn = fmaxf(mI[r], mx[r]);
      alpha[r] = __expf(mI[r] - mn);
      mI[r] = mn;
      lI[r] *= alpha[r];
      rs[r] = 0.f;
    }
#pragma unroll
    for (int t = 0; t < 4; ++t)
#pragma unroll
      for (int r = 0; r < 8; ++r) {
        float p = __expf(s[t][r] * sc - mI[r]);
        s[t][r] = p;
        rs[r] += p;
      }
#pragma unroll
    for (int r = 0; r < 8; ++r) {
      float v0 = rs[r];
      v0 += __shfl_xor(v0, 1, 32);
      v0 += __shfl_xor(v0, 2, 32);
      v0 += __shfl_xor(v0, 4, 32);
      v0 += __shfl_xor(v0, 8, 32);
      lI[r] += v0;
    }
#pragma unroll
    for (int j = 0; j < 9; ++j)
#pragma unroll
      for (int r = 0; r < 8; ++r) acc[j][r] *= alpha[r];

    // re-fragment P (C/D layout -> A layout) via per-wave LDS staging
#pragma unroll
    for (int t = 0; t < 4; ++t)
#pragma unroll
      for (int r = 0; r < 8; ++r)
        myp[(r + 8 * hi) * KT_ + t * 16 + nlo] = (half_t)s[t][r];
    __builtin_amdgcn_wave_barrier();   // same-wave DS ordering is in-order

    v16h pa[2];
#pragma unroll
    for (int ks = 0; ks < 2; ++ks)
      pa[ks] = load_frag_a(myp + ks * 32, KT_);

    // O += P Vext : 9 N-tiles x 2 K-steps = 18 WMMA
#pragma unroll
    for (int j = 0; j < 9; ++j)
#pragma unroll
      for (int ks = 0; ks < 2; ++ks)
        acc[j] = wmma16(pa[ks], load_frag_b(vbT + j * 16 * KT_ + ks * 32, KT_),
                        acc[j]);

    __syncthreads();   // all waves done with buf[stage] before its reuse
    stage ^= 1;
  }

  // epilogue: divide by l, scatter to NCHW outputs
  float inv[8];
#pragma unroll
  for (int r = 0; r < 8; ++r) inv[r] = 1.f / lI[r];

  const size_t rf0Base = (size_t)B_ * C_ * N_;
  const size_t rf1Base = rf0Base + (size_t)B_ * 2 * N_;

#pragma unroll
  for (int j = 0; j < 8; ++j) {
    int c = j * 16 + nlo;
#pragma unroll
    for (int r = 0; r < 8; ++r) {
      int n = qbase + r + 8 * hi;
      out[((size_t)bidx * C_ + c) * N_ + n] = acc[j][r] * inv[r];
    }
  }
  if (nlo < 4) {   // V-tile 8: cols 0,1 -> rf0 ; cols 2,3 -> rf1
    size_t base = (nlo < 2) ? rf0Base : rf1Base;
    int ch = nlo & 1;
#pragma unroll
    for (int r = 0; r < 8; ++r) {
      int n = qbase + r + 8 * hi;
      out[base + ((size_t)bidx * 2 + ch) * N_ + n] = acc[8][r] * inv[r];
    }
  }
}

// ---------------------------------------------------------------------------
extern "C" void kernel_launch(void* const* d_in, const int* in_sizes, int n_in,
                              void* d_out, int out_size, void* d_ws,
                              size_t ws_size, hipStream_t stream) {
  const float* feat  = (const float*)d_in[0];
  const float* flow0 = (const float*)d_in[1];
  const float* flow1 = (const float*)d_in[2];
  const float* Wq    = (const float*)d_in[3];
  const float* bq    = (const float*)d_in[4];
  const float* Wk    = (const float*)d_in[5];
  const float* bk    = (const float*)d_in[6];
  const float* Wv    = (const float*)d_in[7];
  const float* bv    = (const float*)d_in[8];
  float* out = (float*)d_out;

  // workspace (f16): Wq|Wk|Wv (3*16K), qh, kh (B*N*128), vxT (B*160*144*64)
  half_t* wh = (half_t*)d_ws;
  half_t* qh = wh + 3 * C_ * C_;
  half_t* kh = qh + (size_t)B_ * N_ * C_;
  half_t* vx = kh + (size_t)B_ * N_ * C_;
  // total: ~16.5 MB of d_ws

  sap_wconv<<<(3 * C_ * C_ + 255) / 256, 256, 0, stream>>>(Wq, Wk, Wv, wh);
  sap_proj<<<dim3(N_ / QR_, B_), 128, 0, stream>>>(feat, flow0, flow1, wh, bq,
                                                   bk, bv, qh, kh, vx);
  sap_flash<<<dim3(N_ / QR_, B_), 128, 0, stream>>>(qh, kh, vx, out);
}
